// MultiHeadDiffAttention_15805479649307
// MI455X (gfx1250) — compile-verified, matrix operand order checked by reading, not run on hardware
//
#include <hip/hip_runtime.h>

// ---------------------------------------------------------------------------
// MI455X / gfx1250 differential-attention pipeline.
// All matmuls via v_wmma_f32_16x16x32_bf16 (wave32); GEMM tiles staged with
// async global->LDS copies (ASYNCcnt) and double-buffered LDS.
// ---------------------------------------------------------------------------

typedef __bf16 bf16_t;
typedef __attribute__((ext_vector_type(16))) __bf16 v16bf;
typedef __attribute__((ext_vector_type(8)))  __bf16 v8bf;
typedef __attribute__((ext_vector_type(8)))  float  v8f;
typedef __attribute__((ext_vector_type(4)))  int    v4i;

static constexpr float kLambdaInit = 0.35550907f;           // 0.8 - 0.6*exp(-0.3)
static constexpr float kOutScale   = 1.0f - kLambdaInit;
static constexpr float kLnEps      = 1e-5f;
static constexpr int   kB = 4, kT = 1024, kC = 1024, kH = 16, kHs = 64;
static constexpr int   kProjLd = 6144;   // [q1 | q2 | k1 | k2 | v] columns
static constexpr int   kYLd    = 2048;   // concat head outputs (2*hs per head)

__device__ __forceinline__ v8f wmma_bf16(v16bf a, v16bf b, v8f c) {
  return __builtin_amdgcn_wmma_f32_16x16x32_bf16(
      /*neg_a=*/false, a, /*neg_b=*/false, b,
      /*c_mod=*/(short)0, c, /*reuse_a=*/false, /*reuse_b=*/false);
}

// 16-bit A/B fragment striping: lanes 0-15 hold K={0..7,16..23}, lanes 16-31
// hold K={8..15,24..31}; `row` points at this lane's matrix row, kb=(lane>>4)*8.
__device__ __forceinline__ v16bf load_frag(const bf16_t* row, int kb) {
  v8bf lo = *(const v8bf*)(row + kb);
  v8bf hi = *(const v8bf*)(row + 16 + kb);
  return __builtin_shufflevector(lo, hi, 0,1,2,3,4,5,6,7,8,9,10,11,12,13,14,15);
}

// ---- CDNA5 async global->LDS (16B granules), with synchronous fallback ----
#if __has_builtin(__builtin_amdgcn_global_load_async_to_lds_b128)
__device__ __forceinline__ void async_copy16(const bf16_t* g, bf16_t* l) {
  __builtin_amdgcn_global_load_async_to_lds_b128((v4i*)g, (v4i*)l, 0, 0);
}
__device__ __forceinline__ void async_wait0() {
#if __has_builtin(__builtin_amdgcn_s_wait_asynccnt)
  __builtin_amdgcn_s_wait_asynccnt(0);
#else
  asm volatile("s_wait_asynccnt 0x0" ::: "memory");
#endif
}
#else
__device__ __forceinline__ void async_copy16(const bf16_t* g, bf16_t* l) {
  *(v8bf*)l = *(const v8bf*)g;
}
__device__ __forceinline__ void async_wait0() {}
#endif

// ---------------------------------------------------------------------------
// fp32 -> bf16 converters (one-off staging pass; negligible vs GEMMs)
// ---------------------------------------------------------------------------
__global__ void cvt_bf16_kernel(const float* __restrict__ src,
                                bf16_t* __restrict__ dst, int n) {
  int i = blockIdx.x * blockDim.x + threadIdx.x;
  int stride = gridDim.x * blockDim.x;
  for (; i < n; i += stride) dst[i] = (bf16_t)src[i];
}

// dstT[c * dst_ld + r] = src[r * cols + c]  (store weights N-major so GEMM
// B-tiles stage as contiguous 16B copies -> async-to-LDS eligible)
__global__ void cvt_bf16_T_kernel(const float* __restrict__ src,
                                  bf16_t* __restrict__ dstT,
                                  int rows, int cols, int dst_ld) {
  int n = rows * cols;
  int i = blockIdx.x * blockDim.x + threadIdx.x;
  int stride = gridDim.x * blockDim.x;
  for (; i < n; i += stride) {
    int r = i / cols, c = i - r * cols;
    dstT[(size_t)c * dst_ld + r] = (bf16_t)src[i];
  }
}

// ---------------------------------------------------------------------------
// Tiled bf16 GEMM: C(MxN) = A(MxK) * BT(NxK)^T.  Block tile 128x128, BK=32.
// 8 waves as 4(M) x 2(N); wave tile 32x64 = 2x4 WMMA frags (8 WMMA / pass).
// Double-buffered LDS with async global->LDS staging.
// Rows padded to 40 bf16 (80B) to spread b128 fragment reads across banks.
// ---------------------------------------------------------------------------
#define GBM 128
#define GBN 128
#define GBK 32
#define PAD 40

template <bool STORE_F32>
__global__ __launch_bounds__(256)
void gemm_bf16_kernel(const bf16_t* __restrict__ A, const bf16_t* __restrict__ BT,
                      void* __restrict__ C, int M, int N, int K) {
  __shared__ bf16_t As[2][GBM][PAD];
  __shared__ bf16_t Bs[2][GBN][PAD];

  const int tid  = threadIdx.x;
  const int lane = tid & 31;
  const int wave = tid >> 5;
  const int wm   = wave >> 1;       // 0..3
  const int wn   = wave & 1;        // 0..1
  const int bm   = blockIdx.y * GBM;
  const int bn   = blockIdx.x * GBN;
  const int ml   = lane & 15;
  const int kb   = (lane >> 4) * 8;

  // staging: each thread copies 16 A elems + 16 B elems per K-step
  const int sr = tid >> 1;          // 0..127
  const int sc = (tid & 1) * 16;    // 0 or 16

  const bf16_t* ga0 = A  + (size_t)(bm + sr) * K + sc;
  const bf16_t* gb0 = BT + (size_t)(bn + sr) * K + sc;

  v8f acc[2][4] = {};

  // prologue: stage tile 0
  async_copy16(ga0,     &As[0][sr][sc]);
  async_copy16(ga0 + 8, &As[0][sr][sc + 8]);
  async_copy16(gb0,     &Bs[0][sr][sc]);
  async_copy16(gb0 + 8, &Bs[0][sr][sc + 8]);
  async_wait0();
  __syncthreads();

  int buf = 0;
  for (int k0 = 0; k0 < K; k0 += GBK) {
    const int nk = k0 + GBK;
    if (nk < K) {   // stage next tile into the other buffer (async)
      const bf16_t* ga = ga0 + nk;
      const bf16_t* gb = gb0 + nk;
      async_copy16(ga,     &As[buf ^ 1][sr][sc]);
      async_copy16(ga + 8, &As[buf ^ 1][sr][sc + 8]);
      async_copy16(gb,     &Bs[buf ^ 1][sr][sc]);
      async_copy16(gb + 8, &Bs[buf ^ 1][sr][sc + 8]);
    }

    v16bf afr[2], bfr[4];
#pragma unroll
    for (int mi = 0; mi < 2; ++mi)
      afr[mi] = load_frag(&As[buf][wm * 32 + mi * 16 + ml][0], kb);
#pragma unroll
    for (int ni = 0; ni < 4; ++ni)
      bfr[ni] = load_frag(&Bs[buf][wn * 64 + ni * 16 + ml][0], kb);

#pragma unroll
    for (int mi = 0; mi < 2; ++mi)
#pragma unroll
      for (int ni = 0; ni < 4; ++ni)
        acc[mi][ni] = wmma_bf16(afr[mi], bfr[ni], acc[mi][ni]);

    if (nk < K) async_wait0();
    __syncthreads();
    buf ^= 1;
  }

  // epilogue: C/D layout => VGPR r holds M = r + 8*(lane>>4), N = ml
  const int rofs = (lane >> 4) * 8;
#pragma unroll
  for (int mi = 0; mi < 2; ++mi)
#pragma unroll
    for (int ni = 0; ni < 4; ++ni)
#pragma unroll
      for (int r = 0; r < 8; ++r) {
        size_t row = bm + wm * 32 + mi * 16 + rofs + r;
        size_t col = bn + wn * 64 + ni * 16 + ml;
        float v = acc[mi][ni][r];
        if (STORE_F32) ((float*)C)[row * N + col] = v;
        else           ((bf16_t*)C)[row * N + col] = (bf16_t)v;
      }
}

// ---------------------------------------------------------------------------
// Dual-stream flash attention + lambda combine + subLN.
// One block = one (b, h, 128-query-row tile); one wave = 16 query rows.
// ---------------------------------------------------------------------------
__device__ __forceinline__ void stream_update(v8f s[2], float m[8], float l[8],
                                              v8f o[8], bf16_t (*P)[PAD],
                                              const bf16_t (*Vt)[PAD], int lane) {
  const int ml = lane & 15;
  const int kb = (lane >> 4) * 8;
  const int half = lane >> 4;
  float cf[8];

#pragma unroll
  for (int r = 0; r < 8; ++r) {
    float mx = fmaxf(s[0][r], s[1][r]);
#pragma unroll
    for (int off = 1; off < 16; off <<= 1) mx = fmaxf(mx, __shfl_xor(mx, off, 16));
    float mnew = fmaxf(m[r], mx);
    cf[r] = __expf(m[r] - mnew);   // first real step: exp(-inf - v) = 0
    m[r] = mnew;
  }
#pragma unroll
  for (int r = 0; r < 8; ++r) {
    float rs = 0.f;
#pragma unroll
    for (int nt = 0; nt < 2; ++nt) {
      float p = __expf(s[nt][r] - m[r]);
      s[nt][r] = p;
      rs += p;
    }
#pragma unroll
    for (int off = 1; off < 16; off <<= 1) rs += __shfl_xor(rs, off, 16);
    l[r] = l[r] * cf[r] + rs;
  }
  // transpose P through per-wave LDS (C-layout -> A-layout); LDS is in-order
  // per wave so no barrier is required between the store and the load.
#pragma unroll
  for (int nt = 0; nt < 2; ++nt)
#pragma unroll
    for (int r = 0; r < 8; ++r)
      P[half * 8 + r][nt * 16 + ml] = (bf16_t)s[nt][r];

#pragma unroll
  for (int t = 0; t < 8; ++t)
#pragma unroll
    for (int r = 0; r < 8; ++r) o[t][r] *= cf[r];

  v16bf pf = load_frag(&P[ml][0], kb);
#pragma unroll
  for (int t = 0; t < 8; ++t) {
    v16bf vf = load_frag(&Vt[t * 16 + ml][0], kb);
    o[t] = wmma_bf16(pf, vf, o[t]);
  }
}

__global__ __launch_bounds__(256)
void diff_attn_kernel(const bf16_t* __restrict__ proj,   // 4096 x 6144
                      const float* __restrict__ lq1, const float* __restrict__ lk1,
                      const float* __restrict__ lq2, const float* __restrict__ lk2,
                      bf16_t* __restrict__ ybf) {         // 4096 x 2048
  __shared__ bf16_t Vt[128][PAD];       // V tile transposed: [dim][key]
  __shared__ bf16_t Pl[8][16][PAD];     // per-wave P scratch

  const int tid  = threadIdx.x;
  const int lane = tid & 31;
  const int wave = tid >> 5;
  const int ml   = lane & 15;
  const int kb   = (lane >> 4) * 8;
  const int half = lane >> 4;

  const int qt = blockIdx.x & 7;             // 128-row query tile within T
  const int bh = blockIdx.x >> 3;
  const int h  = bh & (kH - 1);
  const int b  = bh >> 4;
  const int rowbase = b * kT;
  const int qbase   = qt * 128 + wave * 16;  // this wave's first query row

  // per-head differential lambda (tiny; computed redundantly per thread)
  float d1 = 0.f, d2 = 0.f;
#pragma unroll 4
  for (int d = 0; d < kHs; ++d) {
    d1 += lq1[h * kHs + d] * lk1[h * kHs + d];
    d2 += lq2[h * kHs + d] * lk2[h * kHs + d];
  }
  const float lam = __expf(d1) - __expf(d2) + kLambdaInit;

  const int q1c = h * kHs;
  const int q2c = kC + h * kHs;
  const int k1c = 2 * kC + h * kHs;
  const int k2c = 3 * kC + h * kHs;
  const int vc  = 4 * kC + h * 128;

  // Q fragments (A-layout: lane = query row, elements = hs), loaded once.
  const bf16_t* rowQ = proj + (size_t)(rowbase + qbase + ml) * kProjLd;
  v16bf q1f[2], q2f[2];
#pragma unroll
  for (int c = 0; c < 2; ++c) {
    q1f[c] = load_frag(rowQ + q1c + c * 32, kb);
    q2f[c] = load_frag(rowQ + q2c + c * 32, kb);
  }

  float m1[8], l1[8], m2[8], l2[8];
  v8f o1[8] = {}, o2[8] = {};
#pragma unroll
  for (int r = 0; r < 8; ++r) {
    m1[r] = -__builtin_inff(); l1[r] = 0.f;
    m2[r] = -__builtin_inff(); l2[r] = 0.f;
  }

  const int kmax = qt * 128 + 128;     // causal upper bound for this block
  for (int k0 = 0; k0 < kmax; k0 += 32) {
    // stage V tile transposed: V[k0+key][vc+dim] -> Vt[dim][key]
#pragma unroll
    for (int p = 0; p < 2; ++p) {
      int key = p * 16 + (tid >> 4);
      int d0  = (tid & 15) * 8;
      v8bf v = *(const v8bf*)(proj + (size_t)(rowbase + k0 + key) * kProjLd + vc + d0);
#pragma unroll
      for (int e = 0; e < 8; ++e) Vt[d0 + e][key] = v[e];
    }
    __syncthreads();

    if (k0 <= qbase + 15) {            // wave has unmasked keys this step
      v8f s1[2] = {}, s2[2] = {};
#pragma unroll
      for (int nt = 0; nt < 2; ++nt) {
        const bf16_t* rowK = proj + (size_t)(rowbase + k0 + nt * 16 + ml) * kProjLd;
#pragma unroll
        for (int c = 0; c < 2; ++c) {
          s1[nt] = wmma_bf16(q1f[c], load_frag(rowK + k1c + c * 32, kb), s1[nt]);
          s2[nt] = wmma_bf16(q2f[c], load_frag(rowK + k2c + c * 32, kb), s2[nt]);
        }
      }
      // scale + causal mask (-1e30 keeps fully-masked rows NaN-free)
#pragma unroll
      for (int nt = 0; nt < 2; ++nt)
#pragma unroll
        for (int r = 0; r < 8; ++r) {
          int q   = qbase + half * 8 + r;
          int key = k0 + nt * 16 + ml;
          bool ok = key <= q;
          s1[nt][r] = ok ? s1[nt][r] * 0.125f : -1.0e30f;
          s2[nt][r] = ok ? s2[nt][r] * 0.125f : -1.0e30f;
        }
      stream_update(s1, m1, l1, o1, Pl[wave], Vt, lane);
      stream_update(s2, m2, l2, o2, Pl[wave], Vt, lane);
    }
    __syncthreads();
  }

  // ---- finalize: y = a1@V - lam * a2@V, then subLN over 128 dims ---------
  float inv1[8], inv2[8];
#pragma unroll
  for (int r = 0; r < 8; ++r) { inv1[r] = 1.0f / l1[r]; inv2[r] = 1.0f / l2[r]; }
  v8f y[8];
#pragma unroll
  for (int t = 0; t < 8; ++t)
#pragma unroll
    for (int r = 0; r < 8; ++r)
      y[t][r] = o1[t][r] * inv1[r] - lam * o2[t][r] * inv2[r];

#pragma unroll
  for (int r = 0; r < 8; ++r) {
    float s = 0.f;
#pragma unroll
    for (int t = 0; t < 8; ++t) s += y[t][r];
#pragma unroll
    for (int off = 1; off < 16; off <<= 1) s += __shfl_xor(s, off, 16);
    float mu = s * (1.0f / 128.0f);

    float v = 0.f;
#pragma unroll
    for (int t = 0; t < 8; ++t) { float d = y[t][r] - mu; v += d * d; }
#pragma unroll
    for (int off = 1; off < 16; off <<= 1) v += __shfl_xor(v, off, 16);
    float rstd = rsqrtf(v * (1.0f / 128.0f) + kLnEps);

    size_t row = (size_t)(b * kT + qbase + half * 8 + r);
#pragma unroll
    for (int t = 0; t < 8; ++t) {
      float val = (y[t][r] - mu) * rstd * kOutScale;
      ybf[row * kYLd + h * 128 + t * 16 + ml] = (bf16_t)val;
    }
  }
}

// ---------------------------------------------------------------------------
// Host orchestration (graph-capture safe).  Scratch (>= 88 MiB in d_ws):
//   [ 0,  8M) xbf   : x bf16                         (4096 x 1024)
//   [ 8, 20M) WbigT : [Wq1|Wq2|Wk1|Wk2|Wv]^T bf16    (6144 x 1024, N-major)
//   [20, 24M) WcT   : Wc^T bf16                      (1024 x 2048, N-major)
//   [24, 72M) proj  : x @ Wbig bf16                  (4096 x 6144)
//   [72, 88M) ybf   : attention output bf16          (4096 x 2048)
// ---------------------------------------------------------------------------
extern "C" void kernel_launch(void* const* d_in, const int* in_sizes, int n_in,
                              void* d_out, int out_size, void* d_ws, size_t ws_size,
                              hipStream_t stream) {
  (void)in_sizes; (void)n_in; (void)out_size; (void)ws_size;
  const float* x   = (const float*)d_in[0];
  const float* Wq1 = (const float*)d_in[1];
  const float* Wq2 = (const float*)d_in[2];
  const float* Wk1 = (const float*)d_in[3];
  const float* Wk2 = (const float*)d_in[4];
  const float* Wv  = (const float*)d_in[5];
  const float* Wc  = (const float*)d_in[6];
  const float* lq1 = (const float*)d_in[7];
  const float* lk1 = (const float*)d_in[8];
  const float* lq2 = (const float*)d_in[9];
  const float* lk2 = (const float*)d_in[10];
  float* out = (float*)d_out;

  char* ws = (char*)d_ws;
  bf16_t* xbf   = (bf16_t*)(ws);
  bf16_t* WbigT = (bf16_t*)(ws + (size_t)8  * 1024 * 1024);
  bf16_t* WcT   = (bf16_t*)(ws + (size_t)20 * 1024 * 1024);
  bf16_t* proj  = (bf16_t*)(ws + (size_t)24 * 1024 * 1024);
  bf16_t* ybf   = (bf16_t*)(ws + (size_t)72 * 1024 * 1024);

  const int rows = kB * kT;  // 4096

  // fp32 -> bf16 staging (weights stored transposed / N-major)
  cvt_bf16_kernel<<<4096, 256, 0, stream>>>(x, xbf, rows * kC);
  cvt_bf16_T_kernel<<<2048, 256, 0, stream>>>(Wq1, WbigT + (size_t)0 * kC * kC, kC, kC, kC);
  cvt_bf16_T_kernel<<<2048, 256, 0, stream>>>(Wq2, WbigT + (size_t)1 * kC * kC, kC, kC, kC);
  cvt_bf16_T_kernel<<<2048, 256, 0, stream>>>(Wk1, WbigT + (size_t)2 * kC * kC, kC, kC, kC);
  cvt_bf16_T_kernel<<<2048, 256, 0, stream>>>(Wk2, WbigT + (size_t)3 * kC * kC, kC, kC, kC);
  cvt_bf16_T_kernel<<<2048, 256, 0, stream>>>(Wv,  WbigT + (size_t)4 * kC * kC, kC, 2 * kC, kC);
  cvt_bf16_T_kernel<<<2048, 256, 0, stream>>>(Wc,  WcT, 2 * kC, kC, 2 * kC);

  // fused projection GEMM: proj = xbf (4096x1024) @ WbigT^T (1024x6144)
  {
    dim3 grid(kProjLd / GBN, rows / GBM);
    gemm_bf16_kernel<false><<<grid, 256, 0, stream>>>(xbf, WbigT, (void*)proj,
                                                      rows, kProjLd, kC);
  }

  // dual-stream causal attention + lambda combine + subLN
  diff_attn_kernel<<<kB * kH * (kT / 128), 256, 0, stream>>>(proj, lq1, lk1, lq2, lk2, ybf);

  // out = ybf (4096x2048) @ WcT^T (2048x1024), fp32 epilogue
  {
    dim3 grid(kC / GBN, rows / GBM);
    gemm_bf16_kernel<true><<<grid, 256, 0, stream>>>(ybf, WcT, (void*)out,
                                                     rows, kC, 2 * kC);
  }
}
